// GRU_Seq2Seq_61048665145594
// MI455X (gfx1250) — matrix operand order repaired and checked by reading, not compiled, not run on hardware
//
#include <hip/hip_runtime.h>
#include <hip/hip_bf16.h>
#include <math.h>

// Problem constants (from reference): B=64, N=128, TS=48, TT=12, C=3, H=128, L=2
#define NBROW 8192   // N*B fused batch
#define HDIM  128
#define TSRC  48
#define TTGT  12
#define CDIM  3
#define MT    64     // rows per block
#define NBLK  (NBROW / MT)   // 128 blocks
#define NTHR  256            // 8 waves (wave32)

#define AS_G __attribute__((address_space(1)))   // global
#define AS_L __attribute__((address_space(3)))   // LDS

typedef __bf16 bf16_t;
typedef __attribute__((ext_vector_type(16))) __bf16 v16bf;
typedef __attribute__((ext_vector_type(8)))  __bf16 v8bf;
typedef __attribute__((ext_vector_type(8)))  float  v8f;
typedef __attribute__((ext_vector_type(4)))  unsigned int u32x4;

__device__ __forceinline__ v8f wmma_bf16(v16bf a, v16bf b, v8f c) {
  // D(f32 16x16) = A(16x32 bf16) * B(32x16 bf16) + C
  return __builtin_amdgcn_wmma_f32_16x16x32_bf16(false, a, false, b, (short)0, c,
                                                 false, false);
}

// A fragment (16x32, MxK) from LDS, row-major [.,lda] bf16.
// Per ISA 7.12.2: lanes 0-15 hold rows m0..m0+15 with K={kb..kb+7, kb+16..kb+23},
// lanes 16-31 hold K={kb+8..kb+15, kb+24..kb+31}.  Pure ds_load_b128 path.
__device__ __forceinline__ v16bf load_a_lds(const AS_L bf16_t* a, int lda, int m0, int kb) {
  const int l   = threadIdx.x & 31;
  const int r   = m0 + (l & 15);
  const int off = (l >> 4) * 8;
  union { v16bf v; v8bf h[2]; } u;
  u.h[0] = *(const AS_L v8bf*)(a + r * lda + kb + off);
  u.h[1] = *(const AS_L v8bf*)(a + r * lda + kb + 16 + off);
  return u.v;
}

// B fragment (32x16, KxN) where B[k][n] = W[n][k], W row-major [.,ldk] bf16 in global.
// Lane l: n = n0 + (l&15), K = {kb+8*(l>>4)..+7, kb+16+8*(l>>4)..+7} -> two global_load_b128 per lane.
__device__ __forceinline__ v16bf load_b_glb(const AS_G bf16_t* w, int ldk, int n0, int kb) {
  const int l  = threadIdx.x & 31;
  const int n  = n0 + (l & 15);
  const int kk = kb + (l >> 4) * 8;
  union { v16bf v; v8bf h[2]; } u;
  const AS_G bf16_t* p = w + n * ldk + kk;
  u.h[0] = *(const AS_G v8bf*)(p);
  u.h[1] = *(const AS_G v8bf*)(p + 16);
  return u.v;
}

// Fast transcendentals: v_exp_f32 + v_rcp_f32, no IEEE divide expansion.
__device__ __forceinline__ float fast_rcp(float x) { return __builtin_amdgcn_rcpf(x); }
__device__ __forceinline__ float sigmf(float x) {
  return fast_rcp(1.f + __expf(-x));
}
__device__ __forceinline__ float fast_tanh(float x) {
  x = fminf(fmaxf(x, -15.f), 15.f);      // keep exp finite
  const float e = __expf(2.f * x);
  return (e - 1.f) * fast_rcp(e + 1.f);
}

// One GRU cell step for a [MT, HDIM] tile of rows, state updated in place (LDS).
// r/z accumulate both x@Wih and h@Whh GEMMs; i_n and h_n kept separate (n-gate gating).
// HAS_GI : dense bf16 input for the gi GEMM (K=HDIM); HAS_SC : scalar C=3 input path.
// NOTE: mt/kb loops deliberately NOT unrolled to bound VGPR liveness (no spills).
template <bool HAS_GI, bool HAS_SC>
__device__ __forceinline__ void gru_layer(
    const AS_L bf16_t* gi_in, const AS_G bf16_t* wih,
    const AS_L float* sXs, const AS_L float* sWs,
    AS_L bf16_t* sH, const AS_G bf16_t* whh,
    const AS_G float* b_ih, const AS_G float* b_hh) {
  const int w    = threadIdx.x >> 5;   // wave id == output column tile (8 waves, 8 tiles)
  const int l    = threadIdx.x & 31;
  const int j0   = w * 16;
  const int n    = j0 + (l & 15);
  const int moff = (l >> 4) * 8;
  const float bir = b_ih[n],            bhr = b_hh[n];
  const float biz = b_ih[HDIM + n],     bhz = b_hh[HDIM + n];
  const float bin = b_ih[2 * HDIM + n], bhn = b_hh[2 * HDIM + n];
  float hnew[4][8];
#pragma unroll 1
  for (int mt = 0; mt < 4; ++mt) {
    const int m0 = mt * 16;
    v8f ar = {}, az = {}, ain = {}, ahn = {};
#pragma unroll 1
    for (int kb = 0; kb < HDIM; kb += 32) {
      v16bf ah = load_a_lds(sH, HDIM, m0, kb);
      ar  = wmma_bf16(ah, load_b_glb(whh, HDIM, 0 * HDIM + j0, kb), ar);
      az  = wmma_bf16(ah, load_b_glb(whh, HDIM, 1 * HDIM + j0, kb), az);
      ahn = wmma_bf16(ah, load_b_glb(whh, HDIM, 2 * HDIM + j0, kb), ahn);
      if constexpr (HAS_GI) {
        v16bf ax = load_a_lds(gi_in, HDIM, m0, kb);
        ar  = wmma_bf16(ax, load_b_glb(wih, HDIM, 0 * HDIM + j0, kb), ar);
        az  = wmma_bf16(ax, load_b_glb(wih, HDIM, 1 * HDIM + j0, kb), az);
        ain = wmma_bf16(ax, load_b_glb(wih, HDIM, 2 * HDIM + j0, kb), ain);
      }
    }
#pragma unroll
    for (int i = 0; i < 8; ++i) {
      const int m = m0 + moff + i;   // C/D layout: VGPR i -> M = i (+8 for lanes>=16)
      float gir = ar[i], giz = az[i], gin = ain[i];
      if constexpr (HAS_SC) {
        const float x0 = sXs[m * 4 + 0], x1 = sXs[m * 4 + 1], x2 = sXs[m * 4 + 2];
        gir += x0 * sWs[(0 * HDIM + n) * 4 + 0] + x1 * sWs[(0 * HDIM + n) * 4 + 1] +
               x2 * sWs[(0 * HDIM + n) * 4 + 2];
        giz += x0 * sWs[(1 * HDIM + n) * 4 + 0] + x1 * sWs[(1 * HDIM + n) * 4 + 1] +
               x2 * sWs[(1 * HDIM + n) * 4 + 2];
        gin += x0 * sWs[(2 * HDIM + n) * 4 + 0] + x1 * sWs[(2 * HDIM + n) * 4 + 1] +
               x2 * sWs[(2 * HDIM + n) * 4 + 2];
      }
      const float r   = sigmf(gir + bir + bhr);
      const float z   = sigmf(giz + biz + bhz);
      const float nnv = fast_tanh(gin + bin + r * (ahn[i] + bhn));
      const float ho  = (float)sH[m * HDIM + n];
      hnew[mt][i] = (1.f - z) * nnv + z * ho;
    }
  }
  __syncthreads();   // all waves done reading old state
#pragma unroll 1
  for (int mt = 0; mt < 4; ++mt)
#pragma unroll
    for (int i = 0; i < 8; ++i)
      sH[(mt * 16 + moff + i) * HDIM + n] = (bf16_t)hnew[mt][i];
  __syncthreads();
}

__global__ __launch_bounds__(NTHR, 1)
void fused_seq2seq(const float* __restrict__ flow_x_,
                   const float* __restrict__ enc_w_ih0_,  // [384][3] f32
                   const float* __restrict__ enc_b_ih_,   // [2][384]
                   const float* __restrict__ enc_b_hh_,
                   const float* __restrict__ enc_hid_b_,  // [128]
                   const float* __restrict__ dec_w_ih0_,  // [384][131] (cols 0..2 f32)
                   const float* __restrict__ dec_b_ih_,
                   const float* __restrict__ dec_b_hh_,
                   const float* __restrict__ attn_b_,     // [128]
                   const float* __restrict__ v_w_,        // [128]
                   const float* __restrict__ fc_w_,       // [3][259]
                   const float* __restrict__ fc_b_,       // [3]
                   const bf16_t* __restrict__ whh0_, const bf16_t* __restrict__ whh1_,
                   const bf16_t* __restrict__ wih1_, const bf16_t* __restrict__ hidw_,
                   const bf16_t* __restrict__ dwih0c_, const bf16_t* __restrict__ dwih1_,
                   const bf16_t* __restrict__ dwhh0_, const bf16_t* __restrict__ dwhh1_,
                   const bf16_t* __restrict__ aw1_, const bf16_t* __restrict__ aw2_,
                   bf16_t* __restrict__ en_out_,   // [TS][NB][H] bf16
                   bf16_t* __restrict__ en_proj_,  // [TS][NB][H] bf16
                   float* __restrict__ pred_) {    // [B][N][TT][C]
  __shared__ bf16_t sH0[MT * HDIM];       // enc layer0 state / scratch tile
  __shared__ bf16_t sH1[MT * HDIM];       // enc layer1 state
  __shared__ bf16_t sD0[MT * HDIM];       // dec layer0 state
  __shared__ bf16_t sD1[MT * HDIM];       // dec layer1 state
  __shared__ bf16_t sCtx[MT * HDIM];      // attention context (bf16 for WMMA A)
  __shared__ float  sF[MT * HDIM];        // hproj (f32)
  __shared__ float  sScore[TSRC * MT];    // attention scores / weights
  __shared__ float  sX[MT * 4];           // per-step C=3 input / decoder feedback
  __shared__ float  sW0[384 * 4];         // scalar (C-column) input weights
  __shared__ float  sAB[HDIM];            // attn_b staged
  __shared__ float  sVW[HDIM];            // v_w staged

  // Address-space-qualified views (global_load / ds_load codegen, no flat).
  const AS_G float* flow_x    = (const AS_G float*)flow_x_;
  const AS_G float* enc_w_ih0 = (const AS_G float*)enc_w_ih0_;
  const AS_G float* enc_b_ih  = (const AS_G float*)enc_b_ih_;
  const AS_G float* enc_b_hh  = (const AS_G float*)enc_b_hh_;
  const AS_G float* enc_hid_b = (const AS_G float*)enc_hid_b_;
  const AS_G float* dec_w_ih0 = (const AS_G float*)dec_w_ih0_;
  const AS_G float* dec_b_ih  = (const AS_G float*)dec_b_ih_;
  const AS_G float* dec_b_hh  = (const AS_G float*)dec_b_hh_;
  const AS_G float* fc_w      = (const AS_G float*)fc_w_;
  const AS_G float* fc_b      = (const AS_G float*)fc_b_;
  const AS_G bf16_t* whh0 = (const AS_G bf16_t*)whh0_;
  const AS_G bf16_t* whh1 = (const AS_G bf16_t*)whh1_;
  const AS_G bf16_t* wih1 = (const AS_G bf16_t*)wih1_;
  const AS_G bf16_t* hidw = (const AS_G bf16_t*)hidw_;
  const AS_G bf16_t* dwih0c = (const AS_G bf16_t*)dwih0c_;
  const AS_G bf16_t* dwih1  = (const AS_G bf16_t*)dwih1_;
  const AS_G bf16_t* dwhh0  = (const AS_G bf16_t*)dwhh0_;
  const AS_G bf16_t* dwhh1  = (const AS_G bf16_t*)dwhh1_;
  const AS_G bf16_t* aw1 = (const AS_G bf16_t*)aw1_;
  const AS_G bf16_t* aw2 = (const AS_G bf16_t*)aw2_;
  AS_G bf16_t* en_out  = (AS_G bf16_t*)en_out_;
  AS_G bf16_t* en_proj = (AS_G bf16_t*)en_proj_;
  AS_G float*  pred    = (AS_G float*)pred_;
  AS_L bf16_t* lH0 = (AS_L bf16_t*)sH0;
  AS_L bf16_t* lH1 = (AS_L bf16_t*)sH1;
  AS_L bf16_t* lD0 = (AS_L bf16_t*)sD0;
  AS_L bf16_t* lD1 = (AS_L bf16_t*)sD1;
  AS_L bf16_t* lCtx = (AS_L bf16_t*)sCtx;
  AS_L float*  lF  = (AS_L float*)sF;
  AS_L float*  lX  = (AS_L float*)sX;
  AS_L float*  lW0 = (AS_L float*)sW0;

  const int tid     = threadIdx.x;
  const int rowbase = blockIdx.x * MT;
  const int w       = tid >> 5;
  const int l       = tid & 31;
  const int j0      = w * 16;
  const int n       = j0 + (l & 15);
  const int moff    = (l >> 4) * 8;

  for (int p = tid; p < MT * HDIM; p += NTHR) { sH0[p] = (bf16_t)0.f; sH1[p] = (bf16_t)0.f; }
  for (int p = tid; p < 384 * 3; p += NTHR) sW0[(p / 3) * 4 + (p % 3)] = enc_w_ih0[p];
  if (tid < HDIM) { sAB[tid] = attn_b_[tid]; sVW[tid] = v_w_[tid]; }
  __syncthreads();

  // ---------------- Encoder: 2-layer GRU over TS steps ----------------
  for (int t = 0; t < TSRC; ++t) {
    for (int p = tid; p < MT * CDIM; p += NTHR) {
      const int m = p / 3, c = p % 3;
      const int row = rowbase + m;
      const int nn = row >> 6, bb = row & 63;  // row = n*64 + b
      sX[m * 4 + c] = flow_x[(((size_t)bb * 128 + nn) * TSRC + t) * CDIM + c];
    }
    __syncthreads();
    gru_layer<false, true>(lH0, whh0, lX, lW0, lH0, whh0, enc_b_ih, enc_b_hh);
    gru_layer<true, false>(lH0, wih1, lX, lW0, lH1, whh1, enc_b_ih + 384, enc_b_hh + 384);
    {  // store en_out tile (contiguous 16KB) in bf16: ds_load_b128 + global_store_b128
      const AS_L u32x4* s = (const AS_L u32x4*)sH1;
      AS_G u32x4* d = (AS_G u32x4*)(en_out + ((size_t)t * NBROW + rowbase) * HDIM);
      for (int p = tid; p < (MT * HDIM * 2) / 16; p += NTHR) d[p] = s[p];
    }
    __syncthreads();
  }

  // ---------------- hidden = tanh(hT @ enc_hid_w.T + b) -> sD0/sD1 ----------------
#pragma unroll 1
  for (int layer = 0; layer < 2; ++layer) {
    const AS_L bf16_t* src = layer ? lH1 : lH0;
    AS_L bf16_t* dst = layer ? lD1 : lD0;
#pragma unroll 1
    for (int mt = 0; mt < 4; ++mt) {
      v8f acc = {};
#pragma unroll 1
      for (int kb = 0; kb < HDIM; kb += 32)
        acc = wmma_bf16(load_a_lds(src, HDIM, mt * 16, kb),
                        load_b_glb(hidw, HDIM, j0, kb), acc);
#pragma unroll
      for (int i = 0; i < 8; ++i)
        dst[(mt * 16 + moff + i) * HDIM + n] = (bf16_t)fast_tanh(acc[i] + enc_hid_b[n]);
    }
  }
  // swap scalar weights to decoder's (dec_w_ih0 cols 0..2)
  for (int p = tid; p < 384 * 3; p += NTHR)
    sW0[(p / 3) * 4 + (p % 3)] = dec_w_ih0[(p / 3) * 131 + (p % 3)];
  __syncthreads();

  // ---------------- en_proj = en_out @ attn_w[:,H:].T (decoder-invariant) ----------
  for (int t = 0; t < TSRC; ++t) {
    const AS_G u32x4* s = (const AS_G u32x4*)(en_out + ((size_t)t * NBROW + rowbase) * HDIM);
    AS_L u32x4* d = (AS_L u32x4*)sH0;
    for (int p = tid; p < (MT * HDIM * 2) / 16; p += NTHR) d[p] = s[p];
    __syncthreads();
#pragma unroll 1
    for (int mt = 0; mt < 4; ++mt) {
      v8f acc = {};
#pragma unroll 1
      for (int kb = 0; kb < HDIM; kb += 32)
        acc = wmma_bf16(load_a_lds(lH0, HDIM, mt * 16, kb),
                        load_b_glb(aw2, HDIM, j0, kb), acc);
#pragma unroll
      for (int i = 0; i < 8; ++i) {
        const int m = mt * 16 + moff + i;
        en_proj[((size_t)t * NBROW + rowbase + m) * HDIM + n] = (bf16_t)acc[i];
      }
    }
    __syncthreads();
  }

  // ---------------- Decoder: TT steps of attention + 2-layer GRU + FC ----------------
  for (int t = 0; t < TTGT; ++t) {
    // 1. hproj = h_last @ attn_w[:, :H].T -> sF (f32)
#pragma unroll 1
    for (int mt = 0; mt < 4; ++mt) {
      v8f acc = {};
#pragma unroll 1
      for (int kb = 0; kb < HDIM; kb += 32)
        acc = wmma_bf16(load_a_lds(lD1, HDIM, mt * 16, kb),
                        load_b_glb(aw1, HDIM, j0, kb), acc);
#pragma unroll
      for (int i = 0; i < 8; ++i)
        lF[(mt * 16 + moff + i) * HDIM + n] = acc[i];
    }
    __syncthreads();
    // 2. scores[ta][m] = v . tanh(en_proj + hproj + attn_b)  (b128 bf16 reads)
#pragma unroll 1
    for (int p = tid; p < TSRC * MT; p += NTHR) {
      const int ta = p >> 6, m = p & 63;
      const AS_G bf16_t* ep = en_proj + ((size_t)ta * NBROW + rowbase + m) * HDIM;
      float s = 0.f;
#pragma unroll 1
      for (int jb = 0; jb < HDIM; jb += 8) {
        const v8bf ev = *(const AS_G v8bf*)(ep + jb);
#pragma unroll
        for (int q = 0; q < 8; ++q)
          s += fast_tanh((float)ev[q] + sF[m * HDIM + jb + q] + sAB[jb + q]) * sVW[jb + q];
      }
      sScore[ta * MT + m] = s;
    }
    __syncthreads();
    // 3. softmax over ta, per row
    if (tid < MT) {
      float mx = -1e30f;
      for (int ta = 0; ta < TSRC; ++ta) mx = fmaxf(mx, sScore[ta * MT + tid]);
      float sum = 0.f;
      for (int ta = 0; ta < TSRC; ++ta) {
        const float e = __expf(sScore[ta * MT + tid] - mx);
        sScore[ta * MT + tid] = e;
        sum += e;
      }
      const float inv = fast_rcp(sum);
      for (int ta = 0; ta < TSRC; ++ta) sScore[ta * MT + tid] *= inv;
    }
    __syncthreads();
    // 4. context[m][j] = sum_ta a * en_out -> sCtx (bf16); each thread owns 32
    //    contiguous columns of one row -> 4 x b128 loads per (ta).
    {
      const int m   = tid >> 2;
      const int jb0 = (tid & 3) * 32;
      float acc[32];
#pragma unroll
      for (int q = 0; q < 32; ++q) acc[q] = 0.f;
#pragma unroll 1
      for (int ta = 0; ta < TSRC; ++ta) {
        const float a = sScore[ta * MT + m];
        const AS_G v8bf* eo =
            (const AS_G v8bf*)(en_out + ((size_t)ta * NBROW + rowbase + m) * HDIM + jb0);
#pragma unroll
        for (int v = 0; v < 4; ++v) {
          const v8bf ev = eo[v];
#pragma unroll
          for (int q = 0; q < 8; ++q) acc[v * 8 + q] += a * (float)ev[q];
        }
      }
#pragma unroll
      for (int q = 0; q < 32; ++q) sCtx[m * HDIM + jb0 + q] = (bf16_t)acc[q];
    }
    __syncthreads();
    // 5. two GRU cells (x0 = [dec_in | context])
    gru_layer<true, true>(lCtx, dwih0c, lX, lW0, lD0, dwhh0, dec_b_ih, dec_b_hh);
    gru_layer<true, false>(lD0, dwih1, lX, lW0, lD1, dwhh1, dec_b_ih + 384, dec_b_hh + 384);
    // 6. out = [h1, h1, dec_in] @ fc_w.T + fc_b ; feeds back as next dec_in
    float outv = 0.f;
    int m = 0, c = 0;
    if (tid < MT * CDIM) {
      m = tid / 3; c = tid % 3;
      float s = fc_b[c];
      const AS_G float* fw = fc_w + c * 259;
      for (int j = 0; j < HDIM; ++j)
        s += (float)sD1[m * HDIM + j] * (fw[j] + fw[HDIM + j]);
      s += sX[m * 4 + 0] * fw[256] + sX[m * 4 + 1] * fw[257] + sX[m * 4 + 2] * fw[258];
      outv = s;
    }
    __syncthreads();
    if (tid < MT * CDIM) {
      sX[m * 4 + c] = outv;
      const int row = rowbase + m;
      const int nn = row >> 6, bb = row & 63;
      pred[(((size_t)bb * 128 + nn) * TTGT + t) * CDIM + c] = outv;
    }
    __syncthreads();
  }
}

// ---------------- weight fp32 -> bf16 packing kernels ----------------
__global__ void cvt_contig(const float* __restrict__ src, bf16_t* __restrict__ dst, int n) {
  const int i = blockIdx.x * blockDim.x + threadIdx.x;
  if (i < n) dst[i] = (bf16_t)src[i];
}
__global__ void cvt_strided(const float* __restrict__ src, bf16_t* __restrict__ dst,
                            int rows, int cols, int ld, int off) {
  const int i = blockIdx.x * blockDim.x + threadIdx.x;
  if (i < rows * cols) {
    const int r = i / cols, c = i % cols;
    dst[i] = (bf16_t)src[r * ld + off + c];
  }
}

extern "C" void kernel_launch(void* const* d_in, const int* in_sizes, int n_in,
                              void* d_out, int out_size, void* d_ws, size_t ws_size,
                              hipStream_t stream) {
  (void)in_sizes; (void)n_in; (void)out_size; (void)ws_size;
  const float* flow_x    = (const float*)d_in[0];
  const float* flow_y    = (const float*)d_in[1];
  const float* enc_w_ih0 = (const float*)d_in[2];
  const float* enc_w_ih1 = (const float*)d_in[3];
  const float* enc_w_hh  = (const float*)d_in[4];
  const float* enc_b_ih  = (const float*)d_in[5];
  const float* enc_b_hh  = (const float*)d_in[6];
  const float* enc_hid_w = (const float*)d_in[7];
  const float* enc_hid_b = (const float*)d_in[8];
  const float* dec_w_ih0 = (const float*)d_in[9];
  const float* dec_w_ih1 = (const float*)d_in[10];
  const float* dec_w_hh  = (const float*)d_in[11];
  const float* dec_b_ih  = (const float*)d_in[12];
  const float* dec_b_hh  = (const float*)d_in[13];
  const float* attn_w    = (const float*)d_in[14];
  const float* attn_b    = (const float*)d_in[15];
  const float* v_w       = (const float*)d_in[16];
  const float* fc_w      = (const float*)d_in[17];
  const float* fc_b      = (const float*)d_in[18];

  char* ws = (char*)d_ws;
  size_t off = 0;
  auto alloc = [&](size_t bytes) -> void* {
    void* p = ws + off;
    off = (off + bytes + 255) & ~(size_t)255;
    return p;
  };
  bf16_t* whh0   = (bf16_t*)alloc(2 * 384 * 128 * sizeof(bf16_t));  // whh0, whh1 adjacent
  bf16_t* whh1   = whh0 + 384 * 128;
  bf16_t* wih1   = (bf16_t*)alloc(384 * 128 * sizeof(bf16_t));
  bf16_t* dwhh0  = (bf16_t*)alloc(2 * 384 * 128 * sizeof(bf16_t));  // dwhh0, dwhh1 adjacent
  bf16_t* dwhh1  = dwhh0 + 384 * 128;
  bf16_t* dwih1  = (bf16_t*)alloc(384 * 128 * sizeof(bf16_t));
  bf16_t* dwih0c = (bf16_t*)alloc(384 * 128 * sizeof(bf16_t));
  bf16_t* hidw   = (bf16_t*)alloc(128 * 128 * sizeof(bf16_t));
  bf16_t* aw1    = (bf16_t*)alloc(128 * 128 * sizeof(bf16_t));
  bf16_t* aw2    = (bf16_t*)alloc(128 * 128 * sizeof(bf16_t));
  bf16_t* en_out  = (bf16_t*)alloc((size_t)TSRC * NBROW * HDIM * sizeof(bf16_t)); // 96MB
  bf16_t* en_proj = (bf16_t*)alloc((size_t)TSRC * NBROW * HDIM * sizeof(bf16_t)); // 96MB

  const int thr = 256;
  cvt_contig<<<(2 * 384 * 128 + thr - 1) / thr, thr, 0, stream>>>(enc_w_hh, whh0, 2 * 384 * 128);
  cvt_contig<<<(384 * 128 + thr - 1) / thr, thr, 0, stream>>>(enc_w_ih1, wih1, 384 * 128);
  cvt_contig<<<(2 * 384 * 128 + thr - 1) / thr, thr, 0, stream>>>(dec_w_hh, dwhh0, 2 * 384 * 128);
  cvt_contig<<<(384 * 128 + thr - 1) / thr, thr, 0, stream>>>(dec_w_ih1, dwih1, 384 * 128);
  cvt_contig<<<(128 * 128 + thr - 1) / thr, thr, 0, stream>>>(enc_hid_w, hidw, 128 * 128);
  cvt_strided<<<(384 * 128 + thr - 1) / thr, thr, 0, stream>>>(dec_w_ih0, dwih0c, 384, 128, 131, 3);
  cvt_strided<<<(128 * 128 + thr - 1) / thr, thr, 0, stream>>>(attn_w, aw1, 128, 128, 256, 0);
  cvt_strided<<<(128 * 128 + thr - 1) / thr, thr, 0, stream>>>(attn_w, aw2, 128, 128, 256, 128);

  float* pred = (float*)d_out;
  fused_seq2seq<<<NBLK, NTHR, 0, stream>>>(
      flow_x, enc_w_ih0, enc_b_ih, enc_b_hh, enc_hid_b, dec_w_ih0, dec_b_ih, dec_b_hh,
      attn_b, v_w, fc_w, fc_b,
      whh0, whh1, wih1, hidw, dwih0c, dwih1, dwhh0, dwhh1, aw1, aw2,
      en_out, en_proj, pred);

  // tuple output: (pred, flow_y) concatenated flat
  const size_t pred_elems = (size_t)64 * 128 * TTGT * CDIM;  // 294912
  hipMemcpyAsync(pred + pred_elems, flow_y, pred_elems * sizeof(float),
                 hipMemcpyDeviceToDevice, stream);
}
